// LSTMRNN_36575941492743
// MI455X (gfx1250) — matrix operand order repaired and checked by reading, not compiled, not run on hardware
//
#include <hip/hip_runtime.h>

// ---------------------------------------------------------------------------
// Seq2seq LSTM (H=1024, L=2, T=64, V=32000) for MI455X / gfx1250, wave32.
// bf16 WMMA (v_wmma_f32_16x16x32_bf16) with fp32 accumulate.
// GEMM: register-reuse of the streamed W operand (1x bandwidth).
// Recurrent matvec: double-buffered global_load_async_to_lds_b128 pipeline.
// ---------------------------------------------------------------------------

#define H_  1024
#define L_  2
#define V_  32000
#define T_  64
#define G4_ 4096   // 4*H

typedef unsigned short u16;
typedef __attribute__((ext_vector_type(16))) __bf16       bf16x16;
typedef __attribute__((ext_vector_type(8)))  float        f32x8;
typedef __attribute__((ext_vector_type(4)))  unsigned int uvec4;

union BQ { uvec4 q[2]; bf16x16 v; };

__device__ __forceinline__ u16 f2bf(float f) {
  unsigned u = __builtin_bit_cast(unsigned, f);
  unsigned r = u + 0x7fffu + ((u >> 16) & 1u);   // round-to-nearest-even
  return (u16)(r >> 16);
}

// A-operand tile (16x32 bf16), row-major X with leading dim ld.
// lanes 0-15: row = lrow, K 0-15 ; lanes 16-31: row = lrow, K 16-31
__device__ __forceinline__ bf16x16 load_a(const u16* __restrict__ X, int ld,
                                          int row, int k0, int lhalf) {
  BQ a;
  const u16* base = X + (size_t)row * ld + k0 + lhalf * 16;
  a.q[0] = *reinterpret_cast<const uvec4*>(base);
  a.q[1] = *reinterpret_cast<const uvec4*>(base + 8);
  return a.v;
}

// B-operand tile (32x16 bf16) = W[N,K]^T : lane n holds column n, 16 contiguous K.
// lanes 0-15: K k0..k0+15 ; lanes 16-31: K k0+16..k0+31  (32B per lane)
__device__ __forceinline__ bf16x16 load_b(const u16* __restrict__ W, int ld,
                                          int col, int k0, int lhalf) {
  BQ b;
  const u16* base = W + (size_t)col * ld + k0 + lhalf * 16;
  b.q[0] = *reinterpret_cast<const uvec4*>(base);
  b.q[1] = *reinterpret_cast<const uvec4*>(base + 8);
  return b.v;
}

// ---------------------------------------------------------------------------
// GEMM: Y[64,N] = X[64,K]_bf16 @ W[N,K]_bf16^T (+ bias).
// One wave owns one 16-column tile and computes ALL four 16-row M slices,
// so the streamed W tile is loaded once and reused by 4 WMMAs (register reuse).
// blockDim = 128 (4 waves = 4 n-tiles), gridDim.x = N/64.
// ---------------------------------------------------------------------------
__global__ void gemm_bf16(const u16* __restrict__ X, const u16* __restrict__ W,
                          const float* __restrict__ bias, float* __restrict__ Y,
                          int N, int K) {
  const int lane  = threadIdx.x & 31;
  const int lrow  = lane & 15;
  const int lhalf = lane >> 4;
  const int n0  = (blockIdx.x * 4 + (threadIdx.x >> 5)) * 16;
  const int col = n0 + lrow;

  f32x8 acc0 = {}, acc1 = {}, acc2 = {}, acc3 = {};
  #pragma unroll 2
  for (int k0 = 0; k0 < K; k0 += 32) {
    __builtin_prefetch((const void*)(W + (size_t)col * K + k0 + 128), 0, 1);
    bf16x16 b  = load_b(W, K, col, k0, lhalf);
    bf16x16 a0 = load_a(X, K,  0 + lrow, k0, lhalf);
    bf16x16 a1 = load_a(X, K, 16 + lrow, k0, lhalf);
    bf16x16 a2 = load_a(X, K, 32 + lrow, k0, lhalf);
    bf16x16 a3 = load_a(X, K, 48 + lrow, k0, lhalf);
    acc0 = __builtin_amdgcn_wmma_f32_16x16x32_bf16(false, a0, false, b, (short)0, acc0, false, false);
    acc1 = __builtin_amdgcn_wmma_f32_16x16x32_bf16(false, a1, false, b, (short)0, acc1, false, false);
    acc2 = __builtin_amdgcn_wmma_f32_16x16x32_bf16(false, a2, false, b, (short)0, acc2, false, false);
    acc3 = __builtin_amdgcn_wmma_f32_16x16x32_bf16(false, a3, false, b, (short)0, acc3, false, false);
  }
  const float bn = bias ? bias[col] : 0.0f;
  #pragma unroll
  for (int i = 0; i < 8; ++i) {
    int r = lhalf * 8 + i;                 // D layout: half0 -> rows 0-7, half1 -> 8-15
    Y[(size_t)(r)      * N + col] = acc0[i] + bn;
    Y[(size_t)(16 + r) * N + col] = acc1[i] + bn;
    Y[(size_t)(32 + r) * N + col] = acc2[i] + bn;
    Y[(size_t)(48 + r) * N + col] = acc3[i] + bn;
  }
}

// ---------------------------------------------------------------------------
// Recurrent matvec: y[4096] = h[1024] @ Whh^T  (+ x2 @ Wih^T if x2 != null).
// The weight stream (B operand) is staged LDS-side with a double-buffered
// global_load_async_to_lds_b128 pipeline (ASYNCcnt-throttled) so the next
// k-step's 1KB tile is in flight while WMMA consumes the current one.
// blockDim=128 (4 waves, one 16-output tile each), gridDim.x = 4096/64 = 64.
// ---------------------------------------------------------------------------
__global__ void lstm_matvec(const u16* __restrict__ h, const u16* __restrict__ Whh,
                            const u16* __restrict__ x2, const u16* __restrict__ Wih,
                            float* __restrict__ y, int K) {
  __shared__ char smem[4 * 2 * 1024];      // 4 waves x 2 slots x 1KB B tile
  const int wave  = threadIdx.x >> 5;
  const int lane  = threadIdx.x & 31;
  const int lrow  = lane & 15;
  const int lhalf = lane >> 4;
  const int n0  = (blockIdx.x * 4 + wave) * 16;
  const int col = n0 + lrow;

  const int S0 = K / 32;                   // k-steps per matrix
  const int S  = x2 ? 2 * S0 : S0;         // total pipeline steps

  char* mybuf = smem + wave * 2048 + lane * 32;

  auto issue = [&](int s, int slot) {
    const u16* M  = (s < S0) ? Whh : Wih;
    int k0 = ((s < S0) ? s : s - S0) * 32;
    const u16* g = M + (size_t)col * K + k0 + lhalf * 16;   // 32B per lane
    unsigned long long ga = (unsigned long long)(size_t)g;
    unsigned lo = (unsigned)(size_t)(mybuf + slot * 1024);  // LDS byte offset
    asm volatile("global_load_async_to_lds_b128 %0, %1, off"
                 :: "v"(lo), "v"(ga) : "memory");
    asm volatile("global_load_async_to_lds_b128 %0, %1, off offset:16"
                 :: "v"(lo), "v"(ga) : "memory");
  };

  issue(0, 0);
  f32x8 acc = {};
  for (int s = 0; s < S; ++s) {
    const int slot = s & 1;
    if (s + 1 < S) {
      issue(s + 1, slot ^ 1);
      // retire step s (async loads complete in order): <=2 outstanding = s+1 only
      asm volatile("s_wait_asynccnt 0x2" ::: "memory");
    } else {
      asm volatile("s_wait_asynccnt 0x0" ::: "memory");
    }
    BQ b;
    const uvec4* lb = reinterpret_cast<const uvec4*>(mybuf + slot * 1024);
    b.q[0] = lb[0];
    b.q[1] = lb[1];
    const u16* vsrc = (s < S0) ? h : x2;
    int k0 = ((s < S0) ? s : s - S0) * 32;
    bf16x16 a = load_a(vsrc, 0, 0, k0, lhalf);   // vector broadcast into all A rows
    acc = __builtin_amdgcn_wmma_f32_16x16x32_bf16(false, a, false, b.v,
                                                  (short)0, acc, false, false);
  }
  if (lane < 16) y[col] = acc[0];          // D row 0 lives in VGPR0, lanes 0-15
}

// ---------------------------------------------------------------------------
// LSTM cell elementwise update (PyTorch gate order i,f,g,o).
// ---------------------------------------------------------------------------
__global__ void lstm_cell(const float* __restrict__ gA, const float* __restrict__ gB,
                          const float* __restrict__ bih, const float* __restrict__ bhh,
                          float* __restrict__ c, u16* __restrict__ h_out,
                          u16* __restrict__ h_copy) {
  int j = blockIdx.x * blockDim.x + threadIdx.x;
  if (j >= H_) return;
  float ig = gA[j]          + bih[j]          + bhh[j];
  float fg = gA[H_ + j]     + bih[H_ + j]     + bhh[H_ + j];
  float gg = gA[2 * H_ + j] + bih[2 * H_ + j] + bhh[2 * H_ + j];
  float og = gA[3 * H_ + j] + bih[3 * H_ + j] + bhh[3 * H_ + j];
  if (gB) { ig += gB[j]; fg += gB[H_ + j]; gg += gB[2 * H_ + j]; og += gB[3 * H_ + j]; }
  float si = 1.0f / (1.0f + __expf(-ig));
  float sf = 1.0f / (1.0f + __expf(-fg));
  float so = 1.0f / (1.0f + __expf(-og));
  float cn = sf * c[j] + si * tanhf(gg);
  float hn = so * tanhf(cn);
  c[j] = cn;
  u16 hb = f2bf(hn);
  h_out[j] = hb;
  if (h_copy) h_copy[j] = hb;
}

// ---------------------------------------------------------------------------
// Helpers: fp32 -> bf16 cast, embedding gathers, state init.
// ---------------------------------------------------------------------------
__global__ void cast_f32_bf16(const float* __restrict__ in, u16* __restrict__ out, long n) {
  long i = (long)blockIdx.x * blockDim.x + threadIdx.x;
  long stride = (long)gridDim.x * blockDim.x;
  for (; i < n; i += stride) out[i] = f2bf(in[i]);
}

__global__ void gather_cast(const int* __restrict__ tok, const float* __restrict__ emb,
                            u16* __restrict__ out) {
  int idx = blockIdx.x * blockDim.x + threadIdx.x;
  if (idx >= T_ * H_) return;
  int t = idx / H_, i = idx - t * H_;
  out[idx] = f2bf(emb[(size_t)tok[t] * H_ + i]);
}

__global__ void gather_cast_dec(const int* __restrict__ tgt, const int* __restrict__ bos,
                                const float* __restrict__ emb, u16* __restrict__ out) {
  int idx = blockIdx.x * blockDim.x + threadIdx.x;
  if (idx >= T_ * H_) return;
  int t = idx / H_, i = idx - t * H_;
  int token = (t == 0) ? bos[0] : tgt[t - 1];
  out[idx] = f2bf(emb[(size_t)token * H_ + i]);
}

__global__ void init_state(u16* __restrict__ h, float* __restrict__ c, int n) {
  int i = blockIdx.x * blockDim.x + threadIdx.x;
  if (i < n) { h[i] = 0; c[i] = 0.0f; }
}

// ---------------------------------------------------------------------------
// Row-wise log-softmax in place (one block per row, V columns).
// ---------------------------------------------------------------------------
__global__ void log_softmax_rows(float* __restrict__ logits, int Vn) {
  float* row = logits + (size_t)blockIdx.x * Vn;
  const int tid = threadIdx.x;
  const int nw = blockDim.x >> 5;
  __shared__ float sred[32];
  __shared__ float sbc;

  float m = -3.402823466e38f;
  for (int i = tid; i < Vn; i += blockDim.x) m = fmaxf(m, row[i]);
  for (int off = 16; off > 0; off >>= 1) m = fmaxf(m, __shfl_xor(m, off, 32));
  if ((tid & 31) == 0) sred[tid >> 5] = m;
  __syncthreads();
  if (tid == 0) {
    float mm = sred[0];
    for (int w = 1; w < nw; ++w) mm = fmaxf(mm, sred[w]);
    sbc = mm;
  }
  __syncthreads();
  m = sbc;

  float s = 0.0f;
  for (int i = tid; i < Vn; i += blockDim.x) s += __expf(row[i] - m);
  for (int off = 16; off > 0; off >>= 1) s += __shfl_xor(s, off, 32);
  if ((tid & 31) == 0) sred[tid >> 5] = s;
  __syncthreads();
  if (tid == 0) {
    float ss = 0.0f;
    for (int w = 0; w < nw; ++w) ss += sred[w];
    sbc = m + __logf(ss);
  }
  __syncthreads();
  float lse = sbc;
  for (int i = tid; i < Vn; i += blockDim.x) row[i] -= lse;
}

// ---------------------------------------------------------------------------
// Host-side orchestration (all launches on `stream`, graph-capture safe).
// ---------------------------------------------------------------------------
extern "C" void kernel_launch(void* const* d_in, const int* in_sizes, int n_in,
                              void* d_out, int out_size, void* d_ws, size_t ws_size,
                              hipStream_t stream) {
  (void)in_sizes; (void)n_in; (void)out_size; (void)ws_size;

  const int*   in_tok  = (const int*)  d_in[0];
  const int*   tgt_tok = (const int*)  d_in[1];
  const int*   bos     = (const int*)  d_in[2];
  const float* emb_src = (const float*)d_in[3];
  const float* enc_Wih = (const float*)d_in[4];
  const float* enc_Whh = (const float*)d_in[5];
  const float* enc_bih = (const float*)d_in[6];
  const float* enc_bhh = (const float*)d_in[7];
  const float* emb_tgt = (const float*)d_in[8];
  const float* dec_Wih = (const float*)d_in[9];
  const float* dec_Whh = (const float*)d_in[10];
  const float* dec_bih = (const float*)d_in[11];
  const float* dec_bhh = (const float*)d_in[12];
  const float* out_W   = (const float*)d_in[13];
  const float* out_b   = (const float*)d_in[14];
  float* out = (float*)d_out;

  // --- workspace carve-out (256B aligned) ---
  char* wsb = (char*)d_ws;
  size_t o = 0;
  auto alloc = [&](size_t bytes) -> char* {
    o = (o + 255) & ~(size_t)255;
    char* p = wsb + o;
    o += bytes;
    return p;
  };
  const size_t WELEM = (size_t)L_ * G4_ * H_;      // 8,388,608 per LSTM weight tensor
  u16*   encWih_bf = (u16*)  alloc(WELEM * 2);
  u16*   encWhh_bf = (u16*)  alloc(WELEM * 2);
  u16*   decWih_bf = (u16*)  alloc(WELEM * 2);
  u16*   decWhh_bf = (u16*)  alloc(WELEM * 2);
  u16*   outW_bf   = (u16*)  alloc((size_t)V_ * H_ * 2);
  u16*   encx_bf   = (u16*)  alloc((size_t)T_ * H_ * 2);
  u16*   decx_bf   = (u16*)  alloc((size_t)T_ * H_ * 2);
  float* Gih0_enc  = (float*)alloc((size_t)T_ * G4_ * 4);
  float* Gih0_dec  = (float*)alloc((size_t)T_ * G4_ * 4);
  u16*   htop_bf   = (u16*)  alloc((size_t)T_ * H_ * 2);
  u16*   hstate    = (u16*)  alloc((size_t)L_ * H_ * 2);
  float* cstate    = (float*)alloc((size_t)L_ * H_ * 4);
  float* lin0      = (float*)alloc((size_t)G4_ * 4);
  float* lin1      = (float*)alloc((size_t)G4_ * 4);

  // --- one-time (per call) prep: state zero, bf16 weight copies, gathers ---
  init_state<<<(L_ * H_ + 255) / 256, 256, 0, stream>>>(hstate, cstate, L_ * H_);
  cast_f32_bf16<<<2048, 256, 0, stream>>>(enc_Wih, encWih_bf, (long)WELEM);
  cast_f32_bf16<<<2048, 256, 0, stream>>>(enc_Whh, encWhh_bf, (long)WELEM);
  cast_f32_bf16<<<2048, 256, 0, stream>>>(dec_Wih, decWih_bf, (long)WELEM);
  cast_f32_bf16<<<2048, 256, 0, stream>>>(dec_Whh, decWhh_bf, (long)WELEM);
  cast_f32_bf16<<<4096, 256, 0, stream>>>(out_W,   outW_bf,   (long)V_ * H_);
  gather_cast    <<<(T_ * H_ + 255) / 256, 256, 0, stream>>>(in_tok, emb_src, encx_bf);
  gather_cast_dec<<<(T_ * H_ + 255) / 256, 256, 0, stream>>>(tgt_tok, bos, emb_tgt, decx_bf);

  // --- batched layer-0 input GEMMs: Gih0 = X @ Wih[0]^T  (64 x 4096 x 1024) ---
  gemm_bf16<<<G4_ / 64, 128, 0, stream>>>(encx_bf, encWih_bf, nullptr, Gih0_enc, G4_, H_);
  gemm_bf16<<<G4_ / 64, 128, 0, stream>>>(decx_bf, decWih_bf, nullptr, Gih0_dec, G4_, H_);

  // --- sequential recurrence: encoder then decoder (state carries over) ---
  const size_t LSTRIDE = (size_t)G4_ * H_;   // per-layer weight elements
  for (int ph = 0; ph < 2; ++ph) {
    const u16*   Wih_bf = ph ? decWih_bf : encWih_bf;
    const u16*   Whh_bf = ph ? decWhh_bf : encWhh_bf;
    const float* bih    = ph ? dec_bih   : enc_bih;
    const float* bhh    = ph ? dec_bhh   : enc_bhh;
    const float* G0     = ph ? Gih0_dec  : Gih0_enc;
    u16*         htop   = ph ? htop_bf   : nullptr;

    u16*   h0 = hstate;        u16*   h1 = hstate + H_;
    float* c0 = cstate;        float* c1 = cstate + H_;
    const u16* Whh0 = Whh_bf;  const u16* Whh1 = Whh_bf + LSTRIDE;
    const u16* Wih1 = Wih_bf + LSTRIDE;

    for (int t = 0; t < T_; ++t) {
      // layer 0: lin0 = h0 @ Whh0^T ; gates = Gih0[t] + lin0 + b
      lstm_matvec<<<64, 128, 0, stream>>>(h0, Whh0, nullptr, nullptr, lin0, H_);
      lstm_cell<<<(H_ + 255) / 256, 256, 0, stream>>>(
          G0 + (size_t)t * G4_, lin0, bih, bhh, c0, h0, nullptr);
      // layer 1: lin1 = h0_new @ Wih1^T + h1 @ Whh1^T ; gates = lin1 + b
      lstm_matvec<<<64, 128, 0, stream>>>(h1, Whh1, h0, Wih1, lin1, H_);
      lstm_cell<<<(H_ + 255) / 256, 256, 0, stream>>>(
          lin1, nullptr, bih + G4_, bhh + G4_, c1, h1,
          htop ? htop + (size_t)t * H_ : nullptr);
    }
  }

  // --- output projection (64 x 32000 x 1024) + log-softmax, into d_out ---
  gemm_bf16<<<V_ / 64, 128, 0, stream>>>(htop_bf, outW_bf, out_b, out, V_, H_);
  log_softmax_rows<<<T_, 1024, 0, stream>>>(out, V_);
}
